// GCN_model2_32908039422336
// MI455X (gfx1250) — compile-verified, matrix operand order checked by reading, not run on hardware
//
#include <hip/hip_runtime.h>
#include <hip/hip_bf16.h>

// ---------------- problem constants (match reference) ----------------
#define N_NODES   50000
#define N_EDGES   1600000
#define IN_FEAT   256
#define H_FEAT    128
#define OUT_FEAT  64
#define N_CLASSES 8

typedef __attribute__((ext_vector_type(16))) __bf16 v16bf;
typedef __attribute__((ext_vector_type(8)))  float  v8f;

// Self-loops are appended logically: edge e in [0,E) comes from edge_index,
// e in [E, E+N) is the self loop (e-E, e-E).
__device__ __forceinline__ int edge_node(const long long* __restrict__ row, int e) {
    return (e < N_EDGES) ? (int)row[e] : (e - N_EDGES);
}

// ---------------- small utility kernels ----------------
__global__ __launch_bounds__(256) void k_zero(float* __restrict__ p, int n) {
    int i = blockIdx.x * blockDim.x + threadIdx.x;
    if (i < n) p[i] = 0.0f;
}

__global__ __launch_bounds__(256) void k_degree(const long long* __restrict__ edst,
                                                float* __restrict__ deg, int etot) {
    int e = blockIdx.x * blockDim.x + threadIdx.x;
    if (e < etot) atomicAdd(&deg[edge_node(edst, e)], 1.0f);
}

__global__ __launch_bounds__(256) void k_dinv(float* __restrict__ deg, int n) {
    int i = blockIdx.x * blockDim.x + threadIdx.x;
    if (i < n) {
        float d = deg[i];
        deg[i] = (d > 0.0f) ? rsqrtf(d) : 0.0f;
    }
}

__global__ __launch_bounds__(256) void k_norm(const long long* __restrict__ esrc,
                                              const long long* __restrict__ edst,
                                              const float* __restrict__ dinv,
                                              float* __restrict__ nrm, int etot) {
    int e = blockIdx.x * blockDim.x + threadIdx.x;
    if (e < etot) nrm[e] = dinv[edge_node(esrc, e)] * dinv[edge_node(edst, e)];
}

// agg[i, j] = b[j]  (bias pre-load; scatter-add lands on top)
__global__ __launch_bounds__(256) void k_bias_init(float* __restrict__ agg,
                                                   const float* __restrict__ b,
                                                   int total, int F) {
    int i = blockIdx.x * blockDim.x + threadIdx.x;
    if (i < total) agg[i] = b[i % F];
}

// ---------------- pack W[KDIM,NC] (f32) -> bf16 in WMMA B-fragment order ----
// Packed layout: for n-tile nt, K-chunk kc, lane, elem i (contiguous 16):
//   Wp[(((nt*KC)+kc)*32 + lane)*16 + i] = bf16( W[k, nt*16 + (lane&15)] )
//   with g = lane>>4,  k = kc*32 + (i<8 ? g*8+i : 16+g*8+(i-8))
template <int KDIM, int NC>
__global__ __launch_bounds__(256) void k_pack_w(const float* __restrict__ W,
                                                __bf16* __restrict__ Wp) {
    const int KC = KDIM / 32;
    int t = blockIdx.x * blockDim.x + threadIdx.x;
    if (t >= NC * KDIM) return;
    int i    = t & 15;
    int lane = (t >> 4) & 31;
    int kc   = (t >> 9) % KC;
    int nt   = (t >> 9) / KC;
    int g = lane >> 4, n = lane & 15;
    int k = kc * 32 + (i < 8 ? g * 8 + i : 16 + g * 8 + (i - 8));
    Wp[t] = (__bf16)W[(size_t)k * NC + nt * 16 + n];
}

// ---------------- WMMA GEMM: C[M,NC] = A[M,KDIM] @ W[KDIM,NC] ----------------
// One wave computes a 16x32 strip of C (two 16x16 N-tiles sharing the A
// fragment). A converted f32->bf16 inline; B read pre-packed (contiguous
// 32B v16bf per lane). Fully unrolled K loop -> back-to-back v_wmma pairs.
// Fragment layouts per CDNA5 ISA 7.12.2 (wave32):
//   A 16x32 bf16 : lane holds row M = lane&15; half g = lane>>4;
//                  elem i<8 -> K=g*8+i ; elem i>=8 -> K=16+g*8+(i-8)
//   C 16x16 f32  : acc[r] -> (M = r + 8*g, N = lane&15)
template <int KDIM, int NC>
__global__ __launch_bounds__(256) void k_gemm_wmma(const float* __restrict__ A,
                                                   const __bf16* __restrict__ Wp,
                                                   float* __restrict__ C, int M) {
    const int lane  = threadIdx.x & 31;
    const int wave  = threadIdx.x >> 5;
    const int npair = (NC / 16) / 2;                       // N-tile pairs
    const int tile  = blockIdx.x * (blockDim.x >> 5) + wave;
    const int mt    = tile / npair;
    const int np    = tile % npair;
    if (mt * 16 >= M) return;          // uniform per-wave exit (EXEC all-1 for WMMA)
    const int nt0 = np * 2, nt1 = nt0 + 1;

    const int ln15 = lane & 15;
    const int g    = lane >> 4;
    const int KC   = KDIM / 32;

    v8f acc0 = {}, acc1 = {};
    const float*  arow = A + (size_t)(mt * 16 + ln15) * KDIM;
    const __bf16* wp0  = Wp + ((size_t)(nt0 * KC) * 32 + lane) * 16;
    const __bf16* wp1  = Wp + ((size_t)(nt1 * KC) * 32 + lane) * 16;

#pragma unroll
    for (int kc = 0; kc < KC; ++kc) {
        const int kb = kc * 32;
        // ---- A fragment: two contiguous runs of 8 floats -> bf16
        const float4* ap0 = (const float4*)(arow + kb + g * 8);
        const float4* ap1 = (const float4*)(arow + kb + 16 + g * 8);
        float4 a0 = ap0[0], a1 = ap0[1];
        float4 a2 = ap1[0], a3 = ap1[1];
        v16bf af;
        af[0]  = (__bf16)a0.x; af[1]  = (__bf16)a0.y; af[2]  = (__bf16)a0.z; af[3]  = (__bf16)a0.w;
        af[4]  = (__bf16)a1.x; af[5]  = (__bf16)a1.y; af[6]  = (__bf16)a1.z; af[7]  = (__bf16)a1.w;
        af[8]  = (__bf16)a2.x; af[9]  = (__bf16)a2.y; af[10] = (__bf16)a2.z; af[11] = (__bf16)a2.w;
        af[12] = (__bf16)a3.x; af[13] = (__bf16)a3.y; af[14] = (__bf16)a3.z; af[15] = (__bf16)a3.w;

        // ---- B fragments: contiguous 32-byte pre-packed loads
        v16bf b0 = *(const v16bf*)(wp0 + (size_t)kc * 32 * 16);
        v16bf b1 = *(const v16bf*)(wp1 + (size_t)kc * 32 * 16);

        acc0 = __builtin_amdgcn_wmma_f32_16x16x32_bf16(
            false, af, false, b0, (short)0, acc0, false, false);
        acc1 = __builtin_amdgcn_wmma_f32_16x16x32_bf16(
            false, af, false, b1, (short)0, acc1, false, false);
    }

#pragma unroll
    for (int r = 0; r < 8; ++r) {
        float* crow = C + (size_t)(mt * 16 + r + 8 * g) * NC;
        crow[nt0 * 16 + ln15] = acc0[r];
        crow[nt1 * 16 + ln15] = acc1[r];
    }
}

// ---------------- edge scatter: agg[dst] += norm[e] * h[src] ----------------
// One thread per (edge, 4-feature chunk); consecutive lanes cover consecutive
// chunks of the same source row -> coalesced float4 gathers.
template <int F>
__global__ __launch_bounds__(256) void k_aggregate(const float* __restrict__ h,
                                                   const long long* __restrict__ esrc,
                                                   const long long* __restrict__ edst,
                                                   const float* __restrict__ nrm,
                                                   float* __restrict__ agg, int etot) {
    const unsigned chunk = F / 4;
    unsigned t = blockIdx.x * 256u + threadIdx.x;
    unsigned tot = (unsigned)etot * chunk;
    if (t >= tot) return;
    int e = (int)(t / chunk);
    int j = (int)(t % chunk) * 4;

    int s = edge_node(esrc, e);
    int d = edge_node(edst, e);
    float w = nrm[e];

    float4 v = *(const float4*)(h + (size_t)s * F + j);
    float* o = agg + (size_t)d * F + j;
    atomicAdd(o + 0, v.x * w);
    atomicAdd(o + 1, v.y * w);
    atomicAdd(o + 2, v.z * w);
    atomicAdd(o + 3, v.w * w);
}

// ---------------- classifier: out = x @ Wc + bc  (64 -> 8) ----------------
__global__ __launch_bounds__(256) void k_classifier(const float* __restrict__ x,
                                                    const float* __restrict__ Wc,
                                                    const float* __restrict__ bc,
                                                    float* __restrict__ out, int n) {
    __shared__ float sW[OUT_FEAT * N_CLASSES];
    __shared__ float sb[N_CLASSES];
    for (int i = threadIdx.x; i < OUT_FEAT * N_CLASSES; i += blockDim.x) sW[i] = Wc[i];
    if (threadIdx.x < N_CLASSES) sb[threadIdx.x] = bc[threadIdx.x];
    __syncthreads();

    int i = blockIdx.x * blockDim.x + threadIdx.x;
    if (i >= n) return;

    float acc[N_CLASSES];
#pragma unroll
    for (int c = 0; c < N_CLASSES; ++c) acc[c] = sb[c];

    const float* row = x + (size_t)i * OUT_FEAT;
#pragma unroll 4
    for (int k = 0; k < OUT_FEAT; ++k) {
        float xv = row[k];
#pragma unroll
        for (int c = 0; c < N_CLASSES; ++c) acc[c] += xv * sW[k * N_CLASSES + c];
    }
#pragma unroll
    for (int c = 0; c < N_CLASSES; ++c) out[(size_t)i * N_CLASSES + c] = acc[c];
}

// ---------------- launcher ----------------
extern "C" void kernel_launch(void* const* d_in, const int* in_sizes, int n_in,
                              void* d_out, int out_size, void* d_ws, size_t ws_size,
                              hipStream_t stream) {
    (void)in_sizes; (void)n_in; (void)out_size; (void)ws_size;

    const float*     seq  = (const float*)d_in[0];
    const long long* ei   = (const long long*)d_in[1];   // int64 [2, E]
    const long long* esrc = ei;                          // row 0
    const long long* edst = ei + N_EDGES;                // row 1
    const float*     W1   = (const float*)d_in[2];
    const float*     b1   = (const float*)d_in[3];
    const float*     W2   = (const float*)d_in[4];
    const float*     b2   = (const float*)d_in[5];
    const float*     Wc   = (const float*)d_in[6];
    const float*     bc   = (const float*)d_in[7];
    float*           out  = (float*)d_out;

    // workspace carve-out
    char*  ws  = (char*)d_ws;
    size_t off = 0;
    auto carve = [&](size_t bytes) -> char* {
        char* p = ws + off;
        off += (bytes + 255) & ~(size_t)255;
        return p;
    };
    const int ETOT = N_EDGES + N_NODES;
    float*  deg  = (float*)carve((size_t)N_NODES * 4);          // becomes dinv in place
    float*  nrm  = (float*)carve((size_t)ETOT * 4);
    float*  h1   = (float*)carve((size_t)N_NODES * H_FEAT * 4);
    float*  agg1 = (float*)carve((size_t)N_NODES * H_FEAT * 4);
    float*  h2   = (float*)carve((size_t)N_NODES * OUT_FEAT * 4);
    float*  agg2 = (float*)carve((size_t)N_NODES * OUT_FEAT * 4);
    __bf16* W1p  = (__bf16*)carve((size_t)IN_FEAT * H_FEAT * 2);
    __bf16* W2p  = (__bf16*)carve((size_t)H_FEAT * OUT_FEAT * 2);

    auto nblk = [](long long n) { return (int)((n + 255) / 256); };

    // ---- normalization coefficients
    k_zero  <<<nblk(N_NODES), 256, 0, stream>>>(deg, N_NODES);
    k_degree<<<nblk(ETOT),    256, 0, stream>>>(edst, deg, ETOT);
    k_dinv  <<<nblk(N_NODES), 256, 0, stream>>>(deg, N_NODES);
    k_norm  <<<nblk(ETOT),    256, 0, stream>>>(esrc, edst, deg, nrm, ETOT);

    // ---- pack weights into bf16 WMMA B-fragment order (once per call)
    k_pack_w<IN_FEAT, H_FEAT ><<<nblk(IN_FEAT * H_FEAT ), 256, 0, stream>>>(W1, W1p);
    k_pack_w<H_FEAT,  OUT_FEAT><<<nblk(H_FEAT * OUT_FEAT), 256, 0, stream>>>(W2, W2p);

    // ---- layer 1: h1 = seq @ W1 ; agg1 = scatter(norm * h1[src]) + b1
    {
        int tiles = (N_NODES / 16) * ((H_FEAT / 16) / 2);
        k_gemm_wmma<IN_FEAT, H_FEAT><<<(tiles + 7) / 8, 256, 0, stream>>>(seq, W1p, h1, N_NODES);
        k_bias_init<<<nblk((long long)N_NODES * H_FEAT), 256, 0, stream>>>(agg1, b1, N_NODES * H_FEAT, H_FEAT);
        k_aggregate<H_FEAT><<<nblk((long long)ETOT * (H_FEAT / 4)), 256, 0, stream>>>(h1, esrc, edst, nrm, agg1, ETOT);
    }

    // ---- layer 2: h2 = agg1 @ W2 ; agg2 = scatter(norm * h2[src]) + b2
    {
        int tiles = (N_NODES / 16) * ((OUT_FEAT / 16) / 2);
        k_gemm_wmma<H_FEAT, OUT_FEAT><<<(tiles + 7) / 8, 256, 0, stream>>>(agg1, W2p, h2, N_NODES);
        k_bias_init<<<nblk((long long)N_NODES * OUT_FEAT), 256, 0, stream>>>(agg2, b2, N_NODES * OUT_FEAT, OUT_FEAT);
        k_aggregate<OUT_FEAT><<<nblk((long long)ETOT * (OUT_FEAT / 4)), 256, 0, stream>>>(h2, esrc, edst, nrm, agg2, ETOT);
    }

    // ---- classifier head
    k_classifier<<<nblk(N_NODES), 256, 0, stream>>>(agg2, Wc, bc, out, N_NODES);
}
